// ImageMoE_46059229282767
// MI455X (gfx1250) — matrix operand order repaired and checked by reading, not compiled, not run on hardware
//
#include <hip/hip_runtime.h>
#include <hip/hip_bf16.h>
#include <math.h>

// ---------------------------------------------------------------------------
// CDNA5 (gfx1250) wave32 WMMA bf16 implementation of the ImageMoE reference.
// Weights pre-converted to bf16 once; bf16 activations between layers; LDS
// tile staging uses GLOBAL_LOAD_ASYNC_TO_LDS_B128 (ASYNCcnt) where it is a
// pure copy, v_perm_b32 RNE packs on the remaining f32 paths.
// All matmuls: v_wmma_f32_16x16x32_bf16 with f32 accumulation.
// ---------------------------------------------------------------------------

typedef __attribute__((ext_vector_type(16))) __bf16 v16bf;
typedef __attribute__((ext_vector_type(8)))  float  v8f;

#define WMMA_BF16(a, b, c) \
  __builtin_amdgcn_wmma_f32_16x16x32_bf16(false, (a), false, (b), (short)0, (c), false, false)

#if defined(__has_builtin)
#if __has_builtin(__builtin_amdgcn_global_load_async_to_lds_b128)
#define HAVE_ASYNC_LDS 1
#endif
#endif

// two f32 -> packed bf16 pair (RNE); high-half merge in one v_perm_b32
__device__ __forceinline__ unsigned int pack2bf(float lo, float hi) {
  unsigned int ul = __builtin_bit_cast(unsigned int, lo);
  unsigned int uh = __builtin_bit_cast(unsigned int, hi);
  ul += 0x7FFFu + ((ul >> 16) & 1u);
  uh += 0x7FFFu + ((uh >> 16) & 1u);
  return __builtin_amdgcn_perm(uh, ul, 0x07060302u);
}
__device__ __forceinline__ unsigned short f2bf(float f) {
  unsigned int u = __builtin_bit_cast(unsigned int, f);
  u += 0x7FFFu + ((u >> 16) & 1u);
  return (unsigned short)(u >> 16);
}
__device__ __forceinline__ float bflo(unsigned int u) {
  return __builtin_bit_cast(float, u << 16);
}
__device__ __forceinline__ float bfhi(unsigned int u) {
  return __builtin_bit_cast(float, u & 0xFFFF0000u);
}

struct U4 { unsigned int a, b, c, d; };
union FragU { U4 u[2]; v16bf v; };

struct Tile16 { float4 f[4]; };                 // 16 f32 (4x b128)
__device__ __forceinline__ Tile16 ldg16(const float* __restrict__ p) {
  Tile16 t;
  t.f[0] = *(const float4*)(p);
  t.f[1] = *(const float4*)(p + 4);
  t.f[2] = *(const float4*)(p + 8);
  t.f[3] = *(const float4*)(p + 12);
  return t;
}
__device__ __forceinline__ void sts16(unsigned short* d, const Tile16& t) {
  U4 u0 = { pack2bf(t.f[0].x, t.f[0].y), pack2bf(t.f[0].z, t.f[0].w),
            pack2bf(t.f[1].x, t.f[1].y), pack2bf(t.f[1].z, t.f[1].w) };
  U4 u1 = { pack2bf(t.f[2].x, t.f[2].y), pack2bf(t.f[2].z, t.f[2].w),
            pack2bf(t.f[3].x, t.f[3].y), pack2bf(t.f[3].z, t.f[3].w) };
  *(U4*)(d) = u0;
  *(U4*)(d + 8) = u1;
}

// 32-byte global -> LDS copy: async (ASYNCcnt, bypasses VGPRs) if available.
#ifdef HAVE_ASYNC_LDS
typedef int v4i_t __attribute__((vector_size(16)));
typedef v4i_t __attribute__((address_space(1)))* gp4_t;   // global AS pointer
typedef v4i_t __attribute__((address_space(3)))* lp4_t;   // LDS AS pointer
#endif
__device__ __forceinline__ void async_cp16(unsigned short* lds, const unsigned short* g) {
#ifdef HAVE_ASYNC_LDS
  // global: same 64-bit address value; LDS: offset lives in the low 32 bits
  gp4_t gs = (gp4_t)(unsigned long long)g;
  lp4_t ld = (lp4_t)(unsigned int)(unsigned long long)lds;
  __builtin_amdgcn_global_load_async_to_lds_b128(gs,     ld,     0, 0);
  __builtin_amdgcn_global_load_async_to_lds_b128(gs + 1, ld + 1, 0, 0);
#else
  U4 a = *(const U4*)(g);
  U4 b = *(const U4*)(g + 8);
  *(U4*)(lds) = a;
  *(U4*)(lds + 8) = b;
#endif
}
__device__ __forceinline__ void async_wait() {
#ifdef HAVE_ASYNC_LDS
  asm volatile("s_wait_asynccnt 0x0" ::: "memory");
#endif
}

// A fragment (16x32 bf16, M x K): VGPR0-3 K=hi*8.., VGPR4-7 K=16+hi*8..
__device__ __forceinline__ v16bf load_frag_a(const unsigned short* sm, int ld) {
  const int lane = threadIdx.x & 31;
  const int m = lane & 15, hi = lane >> 4;
  FragU f;
  f.u[0] = *(const U4*)(sm + m * ld + hi * 8);
  f.u[1] = *(const U4*)(sm + m * ld + 16 + hi * 8);
  return f.v;
}
// B fragment (32x16 bf16) staged as W-rows (N x K): lane n, VGPR0-3 K=hi*8.., VGPR4-7 K=16+hi*8..
__device__ __forceinline__ v16bf load_frag_b(const unsigned short* sm, int ld) {
  const int lane = threadIdx.x & 31;
  const int n = lane & 15, hi = lane >> 4;
  const unsigned short* p = sm + n * ld + hi * 8;
  FragU f;
  f.u[0] = *(const U4*)(p);
  f.u[1] = *(const U4*)(p + 16);
  return f.v;
}

#define GBM 64
#define GBN 64
#define GBK 64
#define GLD 72   // padded LDS row stride (ushorts): bank skew, 16B aligned

// ---------------------------------------------------------------------------
// Weight f32 -> bf16 conversion (run once per launch)
// ---------------------------------------------------------------------------
__global__ void cvt_bf16_kernel(const float* __restrict__ src,
                                unsigned short* __restrict__ dst, int n) {
  const int i = (blockIdx.x * blockDim.x + threadIdx.x) * 8;
  if (i >= n) return;
  const float4 f0 = *(const float4*)(src + i);
  const float4 f1 = *(const float4*)(src + i + 4);
  U4 u = { pack2bf(f0.x, f0.y), pack2bf(f0.z, f0.w),
           pack2bf(f1.x, f1.y), pack2bf(f1.z, f1.w) };
  *(U4*)(dst + i) = u;
}

// ---------------------------------------------------------------------------
// Generic GEMM: out = epilogue(A[MxK] * W[NxK]^T + bias).  W is bf16.
// ABF: A operand is bf16 (async copy staging) vs f32 (convert while staging).
// modes: 0 = f32 out (+bias), 1 = bf16 out (+bias), 2 = bf16 out relu(+bias),
//        3 = QKV split to bf16 q/k/vT, 4 = f32 comb-scaled write,
//        5 = f32 comb-scaled accumulate.   K%64==0, M%64==0, N%64==0.
// ---------------------------------------------------------------------------
template <bool ABF>
__global__ __launch_bounds__(256) void gemm_kernel(
    const void* __restrict__ Ap, const unsigned short* __restrict__ W,
    const float* __restrict__ bias,
    float* __restrict__ Cf, unsigned short* __restrict__ Cb,
    int M, int N, int K, int mode,
    unsigned short* __restrict__ qb, unsigned short* __restrict__ kb,
    unsigned short* __restrict__ vtb,
    const float* __restrict__ combine, int expert)
{
  __shared__ __align__(16) unsigned short As[2][GBM * GLD];
  __shared__ __align__(16) unsigned short Bs[2][GBN * GLD];
  const int bm = blockIdx.y * GBM;
  const int bn = blockIdx.x * GBN;
  const int tid = threadIdx.x;
  const int wave = tid >> 5, lane = tid & 31;
  const int wm = (wave & 3) * 16, wn = (wave >> 2) * 32;
  const int ln = lane & 15, hi = lane >> 4;
  const int lr = tid >> 2, lc = (tid & 3) * 16;
  const int lds_off = lr * GLD + lc;

  const float*          paf = (const float*)Ap + (size_t)(bm + lr) * K + lc;
  const unsigned short* pab = (const unsigned short*)Ap + (size_t)(bm + lr) * K + lc;
  const unsigned short* pw  = W + (size_t)(bn + lr) * K + lc;

  Tile16 taf;          // f32 A staging regs (only used when !ABF)

  // issue loads for k-tile ks into LDS buffer buf
  auto stage = [&](int ks, int buf) {
    if constexpr (ABF) {
      async_cp16(&As[buf][lds_off], pab + (size_t)ks * GBK);
    } else {
      taf = ldg16(paf + (size_t)ks * GBK);          // regs; converted later
    }
    async_cp16(&Bs[buf][lds_off], pw + (size_t)ks * GBK);
  };
  auto finish_store = [&](int buf) {                // f32 path deferred store
    if constexpr (!ABF) sts16(&As[buf][lds_off], taf);
  };

  // prologue: stage k-tile 0
  stage(0, 0);
  finish_store(0);
  async_wait();
  __syncthreads();

  v8f acc0 = {}, acc1 = {};
  const int ksteps = K >> 6;
  int cur = 0;
  for (int ks = 0; ks < ksteps; ++ks) {
    if (ks + 1 < ksteps) stage(ks + 1, cur ^ 1);    // overlap with WMMA
#pragma unroll
    for (int k2 = 0; k2 < 2; ++k2) {
      v16bf a  = load_frag_a(&As[cur][wm * GLD + k2 * 32], GLD);
      v16bf b0 = load_frag_b(&Bs[cur][wn * GLD + k2 * 32], GLD);
      v16bf b1 = load_frag_b(&Bs[cur][(wn + 16) * GLD + k2 * 32], GLD);
      acc0 = WMMA_BF16(a, b0, acc0);
      acc1 = WMMA_BF16(a, b1, acc1);
    }
    if (ks + 1 < ksteps) finish_store(cur ^ 1);
    async_wait();
    __syncthreads();
    cur ^= 1;
  }

#pragma unroll
  for (int r = 0; r < 8; ++r) {
    const int m = bm + wm + hi * 8 + r;
#pragma unroll
    for (int fg = 0; fg < 2; ++fg) {
      const int col = bn + wn + fg * 16 + ln;
      float v = (fg == 0) ? acc0[r] : acc1[r];
      if (bias) v += bias[col];
      switch (mode) {
        case 1:
          Cb[(size_t)m * N + col] = f2bf(v);
          break;
        case 2:
          Cb[(size_t)m * N + col] = f2bf(fmaxf(v, 0.0f));
          break;
        case 3: {  // QKV split: row m = n*1024 + l; col -> (part, head, d)
          const int part = col >> 9, f = col & 511, h = f >> 6, dd = f & 63;
          const int n = m >> 10, l = m & 1023, nh = n * 8 + h;
          const unsigned short bv = f2bf(v);
          if (part == 0)      qb[((size_t)nh * 1024 + l) * 64 + dd] = bv;
          else if (part == 1) kb[((size_t)nh * 1024 + l) * 64 + dd] = bv;
          else                vtb[((size_t)nh * 64 + dd) * 1024 + l] = bv;  // V^T
        } break;
        case 4:
          Cf[(size_t)m * N + col] = combine[(size_t)m * 8 + expert] * v;
          break;
        case 5:
          Cf[(size_t)m * N + col] += combine[(size_t)m * 8 + expert] * v;
          break;
        default:
          Cf[(size_t)m * N + col] = v;
          break;
      }
    }
  }
}

// ---------------------------------------------------------------------------
// Patch embedding -> bf16 tokens.  tok[b, s, :] = (s==0) ? cls + pos[0]
//                         : patches[b, s-1] @ patch_w^T + patch_b + pos[s]
// patch_w pre-converted to bf16 (pwb). Branchless clamped K=196 tail.
// ---------------------------------------------------------------------------
#define PBK 32
#define PLD 40
__global__ __launch_bounds__(256) void patch_embed_kernel(
    const float* __restrict__ img, const unsigned short* __restrict__ pwb,
    const float* __restrict__ pb, const float* __restrict__ cls,
    const float* __restrict__ pos, unsigned short* __restrict__ tokb)
{
  __shared__ __align__(16) unsigned short As[GBM * PLD];
  __shared__ __align__(16) unsigned short Bs[GBN * PLD];
  const int K = 196;
  const int bm = blockIdx.y * GBM;
  const int bn = blockIdx.x * GBN;
  const int tid = threadIdx.x;
  const int wave = tid >> 5, lane = tid & 31;
  const int wm = (wave & 3) * 16, wn = (wave >> 2) * 32;
  const int ln = lane & 15, hi = lane >> 4;
  const int lr = tid >> 2, lc = (tid & 3) * 8;

  const int m_ld = bm + lr;
  const int b_ld = m_ld >> 10, s_ld = m_ld & 1023;
  const int p_ld = (s_ld == 0) ? 0 : (s_ld - 1);       // clamp; s==0 discarded
  const int gy = p_ld >> 5, gx = p_ld & 31;
  const float* ibase = img + (size_t)b_ld * 448 * 448 + (size_t)gy * 14 * 448 + gx * 14;
  const unsigned short* wbase = pwb + (size_t)(bn + lr) * K;

  v8f acc0 = {}, acc1 = {};
  for (int ks = 0; ks < 7; ++ks) {
    const int k0 = ks * PBK;
#pragma unroll
    for (int i = 0; i < 8; i += 2) {
      const int kk0 = k0 + lc + i, kk1 = kk0 + 1;
      const int kc0 = kk0 < K ? kk0 : K - 1;           // clamped addresses
      const int kc1 = kk1 < K ? kk1 : K - 1;
      float v0 = ibase[(size_t)(kc0 / 14) * 448 + (kc0 % 14)];
      float v1 = ibase[(size_t)(kc1 / 14) * 448 + (kc1 % 14)];
      unsigned int w0 = wbase[kc0];
      unsigned int w1 = wbase[kc1];
      v0 = kk0 < K ? v0 : 0.0f;  v1 = kk1 < K ? v1 : 0.0f;   // select, no branch
      w0 = kk0 < K ? w0 : 0u;    w1 = kk1 < K ? w1 : 0u;
      *(unsigned int*)&As[lr * PLD + lc + i] = pack2bf(v0, v1);
      *(unsigned int*)&Bs[lr * PLD + lc + i] = w0 | (w1 << 16);
    }
    __syncthreads();
    v16bf a  = load_frag_a(&As[wm * PLD], PLD);
    v16bf b0 = load_frag_b(&Bs[wn * PLD], PLD);
    v16bf b1 = load_frag_b(&Bs[(wn + 16) * PLD], PLD);
    acc0 = WMMA_BF16(a, b0, acc0);
    acc1 = WMMA_BF16(a, b1, acc1);
    __syncthreads();
  }

#pragma unroll
  for (int r = 0; r < 8; ++r) {
    const int m = bm + wm + hi * 8 + r;
    const int s = m & 1023;
#pragma unroll
    for (int fg = 0; fg < 2; ++fg) {
      const int col = bn + wn + fg * 16 + ln;
      const float v = (fg == 0) ? acc0[r] : acc1[r];
      const float outv = (s == 0) ? (cls[col] + pos[col])
                                  : (v + pb[col] + pos[(size_t)s * 512 + col]);
      tokb[(size_t)m * 512 + col] = f2bf(outv);
    }
  }
}

// ---------------------------------------------------------------------------
// Fused flash attention, all-bf16 operands: q,k [nh][l][d], vT [nh][d][l].
// Staging = async b128 copies. Output attb bf16 at [(n*1024+l)*512 + h*64+d].
// ---------------------------------------------------------------------------
#define ALD 72
#define SLD 68
__global__ __launch_bounds__(256) void attn_kernel(
    const unsigned short* __restrict__ qb, const unsigned short* __restrict__ kb,
    const unsigned short* __restrict__ vtb, unsigned short* __restrict__ att)
{
  __shared__ __align__(16) unsigned short Qs[64 * ALD];
  __shared__ __align__(16) unsigned short Kc[64 * ALD];
  __shared__ __align__(16) unsigned short Vc[64 * ALD];
  __shared__ __align__(16) unsigned short Pb[64 * ALD];
  __shared__ __align__(16) float Sb[64 * SLD];
  __shared__ float mrow[64], lrow[64], srow[64], red[64 * 4];

  const int nh = blockIdx.y;               // n*8 + h
  const int bm = blockIdx.x * 64;          // sequence row tile
  const int tid = threadIdx.x;
  const int wave = tid >> 5, lane = tid & 31;
  const int wm = (wave & 3) * 16, wn = (wave >> 2) * 32;
  const int ln = lane & 15, hi = lane >> 4;
  const int rr = tid >> 2, part = tid & 3;
  const int c0 = part * 16;

  const unsigned short* Q  = qb + ((size_t)nh * 1024 + bm) * 64;
  const unsigned short* Kg = kb + (size_t)nh * 1024 * 64;
  const unsigned short* Vt = vtb + (size_t)nh * 64 * 1024;

  async_cp16(&Qs[rr * ALD + c0], Q + (size_t)rr * 64 + c0);   // stage Q once
  if (tid < 64) { mrow[tid] = -__builtin_inff(); lrow[tid] = 0.0f; }
  v8f o0 = {}, o1 = {};
  async_wait();
  __syncthreads();

  for (int ch = 0; ch < 16; ++ch) {
    async_cp16(&Kc[rr * ALD + c0], Kg + ((size_t)(ch * 64 + rr)) * 64 + c0);
    async_cp16(&Vc[rr * ALD + c0], Vt + (size_t)rr * 1024 + ch * 64 + c0);
    async_wait();
    __syncthreads();

    // S = Q K^T * 1/sqrt(64)
    v8f s0 = {}, s1 = {};
#pragma unroll
    for (int ks = 0; ks < 2; ++ks) {
      v16bf a  = load_frag_a(&Qs[wm * ALD + ks * 32], ALD);
      v16bf b0 = load_frag_b(&Kc[wn * ALD + ks * 32], ALD);
      v16bf b1 = load_frag_b(&Kc[(wn + 16) * ALD + ks * 32], ALD);
      s0 = WMMA_BF16(a, b0, s0);
      s1 = WMMA_BF16(a, b1, s1);
    }
#pragma unroll
    for (int r = 0; r < 8; ++r) {
      const int m = wm + hi * 8 + r;
      Sb[m * SLD + wn + ln]      = 0.125f * s0[r];
      Sb[m * SLD + wn + 16 + ln] = 0.125f * s1[r];
    }
    __syncthreads();

    // online softmax
    {
      float v = -3.4e38f;
#pragma unroll
      for (int i = 0; i < 4; ++i) {
        const float4 s = *(const float4*)&Sb[rr * SLD + c0 + i * 4];
        v = fmaxf(v, fmaxf(fmaxf(s.x, s.y), fmaxf(s.z, s.w)));
      }
      red[rr * 4 + part] = v;
    }
    __syncthreads();
    if (part == 0) {
      const float cm = fmaxf(fmaxf(red[rr * 4], red[rr * 4 + 1]),
                             fmaxf(red[rr * 4 + 2], red[rr * 4 + 3]));
      const float nm = fmaxf(mrow[rr], cm);
      srow[rr] = __expf(mrow[rr] - nm);
      mrow[rr] = nm;
    }
    __syncthreads();
    {
      const float nm = mrow[rr];
      float sum = 0.0f;
#pragma unroll
      for (int i = 0; i < 4; ++i) {
        const float4 s = *(const float4*)&Sb[rr * SLD + c0 + i * 4];
        const float p0 = __expf(s.x - nm), p1 = __expf(s.y - nm);
        const float p2 = __expf(s.z - nm), p3 = __expf(s.w - nm);
        *(unsigned int*)&Pb[rr * ALD + c0 + i * 4]     = pack2bf(p0, p1);
        *(unsigned int*)&Pb[rr * ALD + c0 + i * 4 + 2] = pack2bf(p2, p3);
        sum += (p0 + p1) + (p2 + p3);
      }
      red[rr * 4 + part] = sum;
    }
    __syncthreads();
    if (part == 0)
      lrow[rr] = lrow[rr] * srow[rr] +
                 red[rr * 4] + red[rr * 4 + 1] + red[rr * 4 + 2] + red[rr * 4 + 3];
    __syncthreads();

    // rescale running O, then O += P @ V
#pragma unroll
    for (int r = 0; r < 8; ++r) {
      const float sc = srow[wm + hi * 8 + r];
      o0[r] *= sc; o1[r] *= sc;
    }
#pragma unroll
    for (int ks = 0; ks < 2; ++ks) {
      v16bf a  = load_frag_a(&Pb[wm * ALD + ks * 32], ALD);
      v16bf b0 = load_frag_b(&Vc[wn * ALD + ks * 32], ALD);
      v16bf b1 = load_frag_b(&Vc[(wn + 16) * ALD + ks * 32], ALD);
      o0 = WMMA_BF16(a, b0, o0);
      o1 = WMMA_BF16(a, b1, o1);
    }
    __syncthreads();
  }

  const int n = nh >> 3, h = nh & 7;
#pragma unroll
  for (int r = 0; r < 8; ++r) {
    const int ml = wm + hi * 8 + r;
    const int l = bm + ml;
    const float inv = 1.0f / lrow[ml];
    unsigned short* dst = att + ((size_t)(n * 1024 + l)) * 512 + h * 64;
    dst[wn + ln]      = f2bf(o0[r] * inv);
    dst[wn + 16 + ln] = f2bf(o1[r] * inv);
  }
}

// ---------------------------------------------------------------------------
// Gating on bf16 activations: softmax(8), top-2, renormalized combine weights
// ---------------------------------------------------------------------------
__global__ void gate_kernel(const unsigned short* __restrict__ xb,
                            const float* __restrict__ gw,
                            const float* __restrict__ gb, float* __restrict__ comb)
{
  const int row = blockIdx.x * blockDim.x + threadIdx.x;
  if (row >= 8192) return;
  const unsigned short* xr = xb + (size_t)row * 512;
  float acc[8];
  for (int e = 0; e < 8; ++e) acc[e] = gb[e];
  for (int kk = 0; kk < 512; kk += 8) {
    const U4 xu = *(const U4*)(xr + kk);
    const float x0 = bflo(xu.a), x1 = bfhi(xu.a), x2 = bflo(xu.b), x3 = bfhi(xu.b);
    const float x4 = bflo(xu.c), x5 = bfhi(xu.c), x6 = bflo(xu.d), x7 = bfhi(xu.d);
#pragma unroll
    for (int e = 0; e < 8; ++e) {
      const float4 w0 = *(const float4*)(gw + (size_t)e * 512 + kk);
      const float4 w1 = *(const float4*)(gw + (size_t)e * 512 + kk + 4);
      acc[e] += x0 * w0.x + x1 * w0.y + x2 * w0.z + x3 * w0.w +
                x4 * w1.x + x5 * w1.y + x6 * w1.z + x7 * w1.w;
    }
  }
  float mx = acc[0];
  for (int e = 1; e < 8; ++e) mx = fmaxf(mx, acc[e]);
  float p[8], sum = 0.0f;
  for (int e = 0; e < 8; ++e) { p[e] = __expf(acc[e] - mx); sum += p[e]; }
  for (int e = 0; e < 8; ++e) p[e] /= sum;
  int i0 = 0;
  for (int e = 1; e < 8; ++e) if (p[e] > p[i0]) i0 = e;
  int i1 = (i0 == 0) ? 1 : 0;
  for (int e = 0; e < 8; ++e) { if (e == i0) continue; if (p[e] > p[i1]) i1 = e; }
  const float t = p[i0] + p[i1];
  for (int e = 0; e < 8; ++e)
    comb[(size_t)row * 8 + e] = (e == i0) ? p[i0] / t : (e == i1) ? p[i1] / t : 0.0f;
}

// ---------------------------------------------------------------------------
// cls head: out[b, j] = f[b*1024, :] . cls_w[j, :] + cls_b[j]   (8 x 512, f32)
// ---------------------------------------------------------------------------
__global__ void cls_kernel(const float* __restrict__ f, const float* __restrict__ cw,
                           const float* __restrict__ cb, float* __restrict__ out)
{
  const int idx = blockIdx.x * blockDim.x + threadIdx.x;
  if (idx >= 8 * 512) return;
  const int b = idx >> 9, j = idx & 511;
  const float4* xr = (const float4*)(f + (size_t)(b * 1024) * 512);
  const float4* w  = (const float4*)(cw + (size_t)j * 512);
  float s = cb[j];
  for (int k = 0; k < 128; ++k) {
    const float4 xv = xr[k], wv = w[k];
    s += xv.x * wv.x + xv.y * wv.y + xv.z * wv.z + xv.w * wv.w;
  }
  out[idx] = s;
}

// ---------------------------------------------------------------------------
// Host-side MoE block driver (launch only; everything on `stream`).
// Weights (bf16): ipw, aiw, aow, w1, w2. Biases stay f32.
// ---------------------------------------------------------------------------
static void run_moe(const void* X, bool x_is_f32,
                    const unsigned short* ipw, const float* ipb,
                    const unsigned short* aiw, const float* aib,
                    const unsigned short* aow, const float* aob,
                    const float* gw, const float* gb,
                    const unsigned short* w1, const float* b1,
                    const unsigned short* w2, const float* b2,
                    unsigned short* xab, unsigned short* q, unsigned short* k,
                    unsigned short* vt, unsigned short* attb,
                    float* comb, unsigned short* Hb, float* fout, hipStream_t st)
{
  const dim3 blk(256);
  // inproj: xab = bf16(X @ ipw^T + ipb)
  if (x_is_f32)
    gemm_kernel<false><<<dim3(8, 128), blk, 0, st>>>(X, ipw, ipb, nullptr, xab,
        8192, 512, 512, 1, nullptr, nullptr, nullptr, nullptr, 0);
  else
    gemm_kernel<true><<<dim3(8, 128), blk, 0, st>>>(X, ipw, ipb, nullptr, xab,
        8192, 512, 512, 1, nullptr, nullptr, nullptr, nullptr, 0);
  // qkv: split to bf16 q / k / vT
  gemm_kernel<true><<<dim3(24, 128), blk, 0, st>>>(xab, aiw, aib, nullptr, nullptr,
      8192, 1536, 512, 3, q, k, vt, nullptr, 0);
  // fused attention -> bf16
  attn_kernel<<<dim3(16, 64), blk, 0, st>>>(q, k, vt, attb);
  // out proj -> bf16 (reuse xab)
  gemm_kernel<true><<<dim3(8, 128), blk, 0, st>>>(attb, aow, aob, nullptr, xab,
      8192, 512, 512, 1, nullptr, nullptr, nullptr, nullptr, 0);
  // gating -> combine weights
  gate_kernel<<<dim3(32), blk, 0, st>>>(xab, gw, gb, comb);
  // dense experts: fout = sum_e comb[:,e] * (relu(xab@w1_e^T+b1_e)@w2_e^T+b2_e)
  for (int e = 0; e < 8; ++e) {
    gemm_kernel<true><<<dim3(4, 128), blk, 0, st>>>(xab, w1 + (size_t)e * 256 * 512,
        b1 + (size_t)e * 256, nullptr, Hb,
        8192, 256, 512, 2, nullptr, nullptr, nullptr, nullptr, 0);
    gemm_kernel<true><<<dim3(8, 128), blk, 0, st>>>(Hb, w2 + (size_t)e * 512 * 256,
        b2 + (size_t)e * 512, fout, nullptr,
        8192, 512, 256, (e == 0) ? 4 : 5, nullptr, nullptr, nullptr, comb, e);
  }
}

extern "C" void kernel_launch(void* const* d_in, const int* in_sizes, int n_in,
                              void* d_out, int out_size, void* d_ws, size_t ws_size,
                              hipStream_t stream)
{
  (void)in_sizes; (void)n_in; (void)out_size; (void)ws_size;
  const float* img      = (const float*)d_in[0];
  const float* patch_w  = (const float*)d_in[1];
  const float* patch_b  = (const float*)d_in[2];
  const float* cls_tok  = (const float*)d_in[3];
  const float* pos_emb  = (const float*)d_in[4];
  const float* m1[12]; for (int i = 0; i < 12; ++i) m1[i] = (const float*)d_in[5 + i];
  const float* m2[12]; for (int i = 0; i < 12; ++i) m2[i] = (const float*)d_in[17 + i];
  const float* vector_w = (const float*)d_in[29];
  const float* vector_b = (const float*)d_in[30];
  const float* cls_w    = (const float*)d_in[31];
  const float* cls_b    = (const float*)d_in[32];

  float* out = (float*)d_out;
  char*  base = (char*)d_ws;
  const size_t MB = 1024 * 1024;
  unsigned short* tokb  = (unsigned short*)(base);            // 8 MB (8192x512)
  unsigned short* xab   = (unsigned short*)(base + 8 * MB);   // 8 MB
  unsigned short* qb    = (unsigned short*)(base + 16 * MB);  // 8 MB
  unsigned short* kb    = (unsigned short*)(base + 24 * MB);  // 8 MB
  unsigned short* vtb   = (unsigned short*)(base + 32 * MB);  // 8 MB
  unsigned short* attbb = (unsigned short*)(base + 40 * MB);  // 8 MB
  unsigned short* Hbb   = (unsigned short*)(base + 48 * MB);  // 4 MB (8192x256)
  float*          fout  = (float*)(base + 52 * MB);           // 16 MB (8192x512)
  float*          comb  = (float*)(base + 68 * MB);           // 256 KB
  unsigned short* wpool = (unsigned short*)(base + 69 * MB);  // bf16 weights

  // carve bf16 weight pool
  unsigned short* pwb = wpool;                 size_t o = 100352;        // 512x196
  unsigned short* w1b[2][5];
  const int wsz[5] = { 262144, 786432, 262144, 1048576, 1048576 };       // ip,ai,ao,e1,e2
  for (int mblk = 0; mblk < 2; ++mblk)
    for (int wi = 0; wi < 5; ++wi) { w1b[mblk][wi] = wpool + o; o += wsz[wi]; }
  unsigned short* vecwb = wpool + o;  o += 262144;                       // 512x512

  const dim3 blk(256);
  auto cvt = [&](const float* s, unsigned short* d, int n) {
    cvt_bf16_kernel<<<dim3((n / 8 + 255) / 256), blk, 0, stream>>>(s, d, n);
  };
  // pre-convert all GEMM weights to bf16 (once per launch)
  cvt(patch_w, pwb, 100352);
  const int widx[5] = { 0, 2, 4, 8, 10 };  // inproj_w, attn_in_w, attn_out_w, e_w1, e_w2
  for (int wi = 0; wi < 5; ++wi) { cvt(m1[widx[wi]], w1b[0][wi], wsz[wi]);
                                   cvt(m2[widx[wi]], w1b[1][wi], wsz[wi]); }
  cvt(vector_w, vecwb, 262144);

  // tokens = patch embed + cls + pos  -> bf16
  patch_embed_kernel<<<dim3(8, 128), blk, 0, stream>>>(img, pwb, patch_b,
                                                       cls_tok, pos_emb, tokb);
  // MoE block 1 (bf16 input) -> f1 (f32) in fout
  run_moe(tokb, false, w1b[0][0], m1[1], w1b[0][1], m1[3], w1b[0][2], m1[5],
          m1[6], m1[7], w1b[0][3], m1[9], w1b[0][4], m1[11],
          xab, qb, kb, vtb, attbb, comb, Hbb, fout, stream);
  // first_vector = f1 @ vector_w^T + vector_b  -> d_out[0 .. 4M)
  gemm_kernel<false><<<dim3(8, 128), blk, 0, stream>>>(fout, vecwb, vector_b, out,
      nullptr, 8192, 512, 512, 0, nullptr, nullptr, nullptr, nullptr, 0);
  // cls_first (uses f1 before fout is recycled)
  cls_kernel<<<dim3(16), blk, 0, stream>>>(fout, cls_w, cls_b, out + 8388608);
  // MoE block 2 on first_vector (f32 in d_out) -> f2 in fout
  run_moe(out, true, w1b[1][0], m2[1], w1b[1][1], m2[3], w1b[1][2], m2[5],
          m2[6], m2[7], w1b[1][3], m2[9], w1b[1][4], m2[11],
          xab, qb, kb, vtb, attbb, comb, Hbb, fout, stream);
  // second_vector -> d_out[4M .. 8M)
  gemm_kernel<false><<<dim3(8, 128), blk, 0, stream>>>(fout, vecwb, vector_b,
      out + 4194304, nullptr, 8192, 512, 512, 0, nullptr, nullptr, nullptr, nullptr, 0);
  // cls_second
  cls_kernel<<<dim3(16), blk, 0, stream>>>(fout, cls_w, cls_b, out + 8388608 + 4096);
}
